// MultiheadAttentionMaskMixin_15032385536030
// MI455X (gfx1250) — matrix operand order repaired and checked by reading, not compiled, not run on hardware
//
#include <hip/hip_runtime.h>
#include <hip/hip_bf16.h>
#include <math.h>

// ---------------------------------------------------------------------------
// CDNA5 (gfx1250) wave32 WMMA types
// ---------------------------------------------------------------------------
typedef __bf16 bf16x16 __attribute__((ext_vector_type(16)));
typedef float  f32x8   __attribute__((ext_vector_type(8)));

union AFrag { bf16x16 v; uint4 u[2]; };

static __device__ __forceinline__ f32x8 wmma_bf16(const bf16x16 a, const bf16x16 b, f32x8 c) {
    // D = A(16x32 bf16) x B(32x16 bf16) + C(16x16 f32)
    return __builtin_amdgcn_wmma_f32_16x16x32_bf16(false, a, false, b, (short)0, c, false, false);
}

static __device__ __forceinline__ f32x8 zero8() {
    f32x8 z;
#pragma unroll
    for (int i = 0; i < 8; ++i) z[i] = 0.0f;
    return z;
}

// Async global->LDS 16B copy (GLOBAL_LOAD_ASYNC_TO_LDS_B128, ASYNCcnt).
// LDS byte offset = low 32 bits of the generic pointer (ISA 10.2 aperture map).
static __device__ __forceinline__ void async_copy16(const void* gptr, void* lptr) {
    uint32_t lds_off = (uint32_t)(uintptr_t)lptr;
    asm volatile("global_load_async_to_lds_b128 %0, %1, off"
                 :: "v"(lds_off), "v"(gptr) : "memory");
}
static __device__ __forceinline__ void wait_async0() {
    asm volatile("s_wait_asynccnt 0x0" ::: "memory");
}

// ---------------------------------------------------------------------------
// Kernel 1/4: C[M,N] = A[M,K] @ W[N,K]^T + bias, optional clip.
// bf16x3 split precision: acc += Ahi*Whi + Ahi*Wlo + Alo*Whi.
// Block: 256 threads = 8 waves (2x4), tile 64(M) x 128(N), K-step 32,
// double-buffered LDS (software pipelined: VMEM for k+1 under WMMA of k).
// ---------------------------------------------------------------------------
#define GTM 64
#define GTN 128
#define GTK 32

__global__ __launch_bounds__(256) void gemm_nt_bf16x3(
    const float* __restrict__ A,   // [M][K]
    const float* __restrict__ W,   // [N][K]
    const float* __restrict__ bias,// [N]
    float* __restrict__ C,         // [M][N]
    int M, int N, int K, float clipv)
{
    __shared__ __align__(16) __bf16 Ahi[2][GTM][GTK];
    __shared__ __align__(16) __bf16 Alo[2][GTM][GTK];
    __shared__ __align__(16) __bf16 Whi[2][GTN][GTK];
    __shared__ __align__(16) __bf16 Wlo[2][GTN][GTK];

    const int tid   = threadIdx.x;
    const int lane  = tid & 31;
    const int wave  = tid >> 5;
    const int wm    = wave >> 2;      // 0..1  (32 M rows each)
    const int wn    = wave & 3;       // 0..3  (32 N cols each)
    const int l15   = lane & 15;
    const int khalf = lane >> 4;
    const int m0    = blockIdx.x * GTM;
    const int n0    = blockIdx.y * GTN;

    const int ar = tid >> 3, ac4 = (tid & 7) * 4;   // A/W staging coords

    f32x8 acc[2][2];
#pragma unroll
    for (int i = 0; i < 2; ++i)
#pragma unroll
        for (int j = 0; j < 2; ++j) acc[i][j] = zero8();

    float4 ra[2], rw[4];
    auto gload = [&](int k0) {
#pragma unroll
        for (int it = 0; it < 2; ++it)
            ra[it] = *reinterpret_cast<const float4*>(
                &A[(size_t)(m0 + ar + it * 32) * K + k0 + ac4]);
#pragma unroll
        for (int it = 0; it < 4; ++it)
            rw[it] = *reinterpret_cast<const float4*>(
                &W[(size_t)(n0 + ar + it * 32) * K + k0 + ac4]);
    };
    auto lstore = [&](int buf) {
#pragma unroll
        for (int it = 0; it < 2; ++it) {
            float v[4] = {ra[it].x, ra[it].y, ra[it].z, ra[it].w};
#pragma unroll
            for (int j = 0; j < 4; ++j) {
                __bf16 h = (__bf16)v[j];
                Ahi[buf][ar + it * 32][ac4 + j] = h;
                Alo[buf][ar + it * 32][ac4 + j] = (__bf16)(v[j] - (float)h);
            }
        }
#pragma unroll
        for (int it = 0; it < 4; ++it) {
            float v[4] = {rw[it].x, rw[it].y, rw[it].z, rw[it].w};
#pragma unroll
            for (int j = 0; j < 4; ++j) {
                __bf16 h = (__bf16)v[j];
                Whi[buf][ar + it * 32][ac4 + j] = h;
                Wlo[buf][ar + it * 32][ac4 + j] = (__bf16)(v[j] - (float)h);
            }
        }
    };

    const int nk = K / GTK;
    gload(0);
    for (int kt = 0; kt < nk; ++kt) {
        const int buf = kt & 1;
        lstore(buf);
        __syncthreads();
        if (kt + 1 < nk) gload((kt + 1) * GTK);           // VMEM under WMMA
        if (kt + 2 < nk) {                                // L2 prefetch k+2
            const int k2 = (kt + 2) * GTK;
            __builtin_prefetch(&A[(size_t)(m0 + ar) * K + k2], 0, 0);
            __builtin_prefetch(&W[(size_t)(n0 + ar) * K + k2], 0, 0);
        }

        // fragment gathers (contiguous 16B LDS reads, ISA 16-bit layouts)
        bf16x16 ah[2], al[2], bh[2], bl[2];
#pragma unroll
        for (int mf = 0; mf < 2; ++mf) {
            int row = wm * 32 + mf * 16 + l15;
            AFrag t0, t1;
            t0.u[0] = *(const uint4*)&Ahi[buf][row][khalf * 8];
            t0.u[1] = *(const uint4*)&Ahi[buf][row][16 + khalf * 8];
            t1.u[0] = *(const uint4*)&Alo[buf][row][khalf * 8];
            t1.u[1] = *(const uint4*)&Alo[buf][row][16 + khalf * 8];
            ah[mf] = t0.v; al[mf] = t1.v;
        }
#pragma unroll
        for (int nf = 0; nf < 2; ++nf) {
            int col = wn * 32 + nf * 16 + l15;
            AFrag t0, t1;
            t0.u[0] = *(const uint4*)&Whi[buf][col][khalf * 8];
            t0.u[1] = *(const uint4*)&Whi[buf][col][16 + khalf * 8];
            t1.u[0] = *(const uint4*)&Wlo[buf][col][khalf * 8];
            t1.u[1] = *(const uint4*)&Wlo[buf][col][16 + khalf * 8];
            bh[nf] = t0.v; bl[nf] = t1.v;
        }
#pragma unroll
        for (int mf = 0; mf < 2; ++mf)
#pragma unroll
            for (int nf = 0; nf < 2; ++nf) {
                acc[mf][nf] = wmma_bf16(ah[mf], bh[nf], acc[mf][nf]);
                acc[mf][nf] = wmma_bf16(ah[mf], bl[nf], acc[mf][nf]);
                acc[mf][nf] = wmma_bf16(al[mf], bh[nf], acc[mf][nf]);
            }
    }

    // epilogue: bias (+ clip) ; C layout: M = r + 8*khalf, N = l15
#pragma unroll
    for (int mf = 0; mf < 2; ++mf)
#pragma unroll
        for (int nf = 0; nf < 2; ++nf) {
            int col = n0 + wn * 32 + nf * 16 + l15;
            float bv = bias[col];
#pragma unroll
            for (int r = 0; r < 8; ++r) {
                int row = m0 + wm * 32 + mf * 16 + khalf * 8 + r;
                float v = acc[mf][nf][r] + bv;
                if (clipv > 0.0f) v = fminf(fmaxf(v, -clipv), clipv);
                C[(size_t)row * N + col] = v;
            }
        }
}

// ---------------------------------------------------------------------------
// Kernel 2: per-token LayerNorm of q and k over D=2048, pack to bf16 heads.
//   qb,kb: [B,H,S,hd] ; vT: [B,H,hd,S] (transposed for PV B-fragments)
// ---------------------------------------------------------------------------
__global__ __launch_bounds__(256) void ln_pack(
    const float* __restrict__ qkv,           // [4096][6144]
    const float* __restrict__ g_q, const float* __restrict__ b_q,
    const float* __restrict__ g_k, const float* __restrict__ b_k,
    __bf16* __restrict__ qb, __bf16* __restrict__ kb, __bf16* __restrict__ vT)
{
    const int t   = blockIdx.x;          // token id 0..4095
    const int b   = t >> 11;
    const int s   = t & 2047;
    const int tid = threadIdx.x;
    const int wave = tid >> 5;
    const int lane = tid & 31;
    __shared__ float red[16];

    for (int which = 0; which < 2; ++which) {
        const float* row = qkv + (size_t)t * 6144 + which * 2048;
        const float* g = which ? g_k : g_q;
        const float* be = which ? b_k : b_q;
        __bf16* dst = which ? kb : qb;

        float vals[8];
        float sum = 0.0f, sq = 0.0f;
#pragma unroll
        for (int i = 0; i < 8; ++i) {
            float x = row[tid + i * 256];
            vals[i] = x; sum += x; sq += x * x;
        }
#pragma unroll
        for (int o = 16; o >= 1; o >>= 1) {
            sum += __shfl_xor(sum, o, 32);
            sq  += __shfl_xor(sq,  o, 32);
        }
        if (lane == 0) { red[wave] = sum; red[8 + wave] = sq; }
        __syncthreads();
        float ts = 0.0f, tq = 0.0f;
#pragma unroll
        for (int w = 0; w < 8; ++w) { ts += red[w]; tq += red[8 + w]; }
        const float mu  = ts * (1.0f / 2048.0f);
        const float var = tq * (1.0f / 2048.0f) - mu * mu;
        const float rs  = rsqrtf(var + 1e-5f);
#pragma unroll
        for (int i = 0; i < 8; ++i) {
            int d = tid + i * 256;
            float y = (vals[i] - mu) * rs * g[d] + be[d];
            int h = d >> 7, dd = d & 127;
            dst[((size_t)(b * 16 + h) * 2048 + s) * 128 + dd] = (__bf16)y;
        }
        __syncthreads();   // red[] reused next iteration
    }

    // v: convert + transpose into [B,H,hd,S]
    const float* vrow = qkv + (size_t)t * 6144 + 4096;
#pragma unroll
    for (int i = 0; i < 8; ++i) {
        int d = tid + i * 256;
        int h = d >> 7, dd = d & 127;
        vT[((size_t)(b * 16 + h) * 128 + dd) * 2048 + s] = (__bf16)vrow[d];
    }
}

// ---------------------------------------------------------------------------
// Kernel 3: flash attention. Block = 64 query rows of one (b,h).
// 128 threads = 4 waves; wave w owns q rows [w*16, w*16+16), full hd=128 acc.
// K/V tiles double-buffered and fetched with async global->LDS DMA.
// ---------------------------------------------------------------------------
__global__ __launch_bounds__(128) void flash_attn(
    const __bf16* __restrict__ qb,   // [BH][2048][128]
    const __bf16* __restrict__ kb,   // [BH][2048][128]
    const __bf16* __restrict__ vT,   // [BH][128][2048]
    const float*  __restrict__ attn_bias,   // [16][2048]
    const unsigned char* __restrict__ amask,// [B][2048]
    const unsigned char* __restrict__ qmask,// [B][2048][2048]
    float* __restrict__ ctx)                // [B][2048][2048]
{
    const int qt   = blockIdx.x;     // query tile 0..31
    const int bh   = blockIdx.y;     // 0..31
    const int b    = bh >> 4, h = bh & 15;
    const int tid  = threadIdx.x;
    const int lane = tid & 31;
    const int wave = tid >> 5;
    const int l15  = lane & 15;
    const int khalf = lane >> 4;
    const int q0   = qt * 64;

    __shared__ __align__(16) __bf16 Qs[64][128];
    __shared__ __align__(16) __bf16 Ks[2][64][128];
    __shared__ __align__(16) __bf16 Vs[2][128][64];   // transposed: [d][kv]
    __shared__ __align__(16) __bf16 Ps[4][16][64];    // per-wave P staging

    // ---- load Q tile (64x128 bf16)
    const __bf16* qbase = qb + ((size_t)bh * 2048 + q0) * 128;
#pragma unroll
    for (int it = 0; it < 8; ++it) {
        int idx = tid + it * 128;
        int r = idx >> 4, c = (idx & 15) * 8;
        *(uint4*)&Qs[r][c] = *(const uint4*)&qbase[(size_t)r * 128 + c];
    }
    __syncthreads();

    // ---- build Q A-fragments (4 d-steps of 32)
    bf16x16 qf[4];
    {
        int row = wave * 16 + l15;
#pragma unroll
        for (int ds = 0; ds < 4; ++ds) {
            AFrag f;
            f.u[0] = *(const uint4*)&Qs[row][ds * 32 + khalf * 8];
            f.u[1] = *(const uint4*)&Qs[row][ds * 32 + 16 + khalf * 8];
            qf[ds] = f.v;
        }
    }

    // async K/V tile fetch: 16KB each -> 8+8 b128 DMAs per thread
    auto issue_tile = [&](int kt, int buf) {
        const int k0 = kt * 64;
        const __bf16* kbase = kb + ((size_t)bh * 2048 + k0) * 128;
#pragma unroll
        for (int it = 0; it < 8; ++it) {
            int idx = tid + it * 128;
            int r = idx >> 4, c = (idx & 15) * 8;
            async_copy16(&kbase[(size_t)r * 128 + c], &Ks[buf][r][c]);
        }
        const __bf16* vbase = vT + (size_t)bh * 128 * 2048 + k0;
#pragma unroll
        for (int it = 0; it < 8; ++it) {
            int idx = tid + it * 128;
            int r = idx >> 3, c = (idx & 7) * 8;
            async_copy16(&vbase[(size_t)r * 2048 + c], &Vs[buf][r][c]);
        }
    };

    float mi[8], li[8];
    f32x8 acc[8];
#pragma unroll
    for (int r = 0; r < 8; ++r) { mi[r] = -INFINITY; li[r] = 0.0f; }
#pragma unroll
    for (int j = 0; j < 8; ++j) acc[j] = zero8();

    const float scale = 0.0883883476483184f;           // 1/sqrt(128)
    const float MINV  = -3.4028234663852886e38f;       // finfo(f32).min

    issue_tile(0, 0);
    for (int kt = 0; kt <= qt; ++kt) {                 // causal tile range
        const int k0 = kt * 64;
        const int buf = kt & 1;
        wait_async0();                                 // my DMAs into buf done
        __syncthreads();                               // everyone's DMAs done
        if (kt + 1 <= qt) issue_tile(kt + 1, buf ^ 1); // DMA k+1 under WMMA k

        // ---- scores S = Q K^T  (4 frags of 16 kv cols)
        f32x8 sc[4];
#pragma unroll
        for (int nf = 0; nf < 4; ++nf) sc[nf] = zero8();
#pragma unroll
        for (int nf = 0; nf < 4; ++nf) {
            int n = nf * 16 + l15;
#pragma unroll
            for (int ds = 0; ds < 4; ++ds) {
                AFrag bf;
                bf.u[0] = *(const uint4*)&Ks[buf][n][ds * 32 + khalf * 8];
                bf.u[1] = *(const uint4*)&Ks[buf][n][ds * 32 + 16 + khalf * 8];
                sc[nf] = wmma_bf16(qf[ds], bf.v, sc[nf]);
            }
        }

        // ---- scale + bias, then masks -> MINV (reference ordering)
#pragma unroll
        for (int nf = 0; nf < 4; ++nf) {
            int kc = k0 + nf * 16 + l15;
            float bv = attn_bias[h * 2048 + kc];
            bool am = amask[b * 2048 + kc] != 0;
            const unsigned char* qmcol = qmask + (size_t)b * 2048 * 2048 + kc;
#pragma unroll
            for (int r = 0; r < 8; ++r) {
                int qr = q0 + wave * 16 + khalf * 8 + r;
                float sv = sc[nf][r] * scale + bv;
                bool ok = am && (qr >= kc) && (qmcol[(size_t)qr * 2048] != 0);
                sc[nf][r] = ok ? sv : MINV;
            }
        }

        // ---- online softmax update (row stats shuffled across 16-lane half)
#pragma unroll
        for (int r = 0; r < 8; ++r) {
            float rm = fmaxf(fmaxf(sc[0][r], sc[1][r]), fmaxf(sc[2][r], sc[3][r]));
            rm = fmaxf(rm, __shfl_xor(rm, 1, 32));
            rm = fmaxf(rm, __shfl_xor(rm, 2, 32));
            rm = fmaxf(rm, __shfl_xor(rm, 4, 32));
            rm = fmaxf(rm, __shfl_xor(rm, 8, 32));
            float mnew  = fmaxf(mi[r], rm);
            float alpha = __expf(mi[r] - mnew);
            mi[r] = mnew;
            float rs = 0.0f;
#pragma unroll
            for (int nf = 0; nf < 4; ++nf) {
                float p = __expf(sc[nf][r] - mnew);
                sc[nf][r] = p; rs += p;
            }
            rs += __shfl_xor(rs, 1, 32);
            rs += __shfl_xor(rs, 2, 32);
            rs += __shfl_xor(rs, 4, 32);
            rs += __shfl_xor(rs, 8, 32);
            li[r] = li[r] * alpha + rs;
#pragma unroll
            for (int j = 0; j < 8; ++j) acc[j][r] *= alpha;
        }

        // ---- restage P (C-layout f32) -> LDS bf16 (A-layout source)
#pragma unroll
        for (int nf = 0; nf < 4; ++nf)
#pragma unroll
            for (int r = 0; r < 8; ++r)
                Ps[wave][khalf * 8 + r][nf * 16 + l15] = (__bf16)sc[nf][r];
        __syncthreads();   // order Ps stores before gathers

        // ---- acc += P @ V   (2 kv-steps of 32, 8 d-frags)
#pragma unroll
        for (int ks = 0; ks < 2; ++ks) {
            AFrag pa;
            pa.u[0] = *(const uint4*)&Ps[wave][l15][ks * 32 + khalf * 8];
            pa.u[1] = *(const uint4*)&Ps[wave][l15][ks * 32 + 16 + khalf * 8];
#pragma unroll
            for (int j = 0; j < 8; ++j) {
                AFrag vb;
                int n = j * 16 + l15;
                vb.u[0] = *(const uint4*)&Vs[buf][n][ks * 32 + khalf * 8];
                vb.u[1] = *(const uint4*)&Vs[buf][n][ks * 32 + 16 + khalf * 8];
                acc[j] = wmma_bf16(pa.v, vb.v, acc[j]);
            }
        }
    }

    // ---- epilogue: ctx[b, qr, h*128 + d] = acc / l
    float* cbase = ctx + (size_t)b * 2048 * 2048;
#pragma unroll
    for (int r = 0; r < 8; ++r) {
        int qr = q0 + wave * 16 + khalf * 8 + r;
        float inv = 1.0f / li[r];
#pragma unroll
        for (int j = 0; j < 8; ++j)
            cbase[(size_t)qr * 2048 + h * 128 + j * 16 + l15] = acc[j][r] * inv;
    }
}

// ---------------------------------------------------------------------------
// Host launcher
// ---------------------------------------------------------------------------
extern "C" void kernel_launch(void* const* d_in, const int* in_sizes, int n_in,
                              void* d_out, int out_size, void* d_ws, size_t ws_size,
                              hipStream_t stream) {
    (void)in_sizes; (void)n_in; (void)out_size; (void)ws_size;
    const float* x       = (const float*)d_in[0];
    const float* Wqkv_w  = (const float*)d_in[1];
    const float* Wqkv_b  = (const float*)d_in[2];
    const float* q_ln_g  = (const float*)d_in[3];
    const float* q_ln_b  = (const float*)d_in[4];
    const float* k_ln_g  = (const float*)d_in[5];
    const float* k_ln_b  = (const float*)d_in[6];
    const float* out_w   = (const float*)d_in[7];
    const float* out_b   = (const float*)d_in[8];
    const float* attn_bias = (const float*)d_in[9];
    const unsigned char* amask = (const unsigned char*)d_in[10];
    const unsigned char* qmask = (const unsigned char*)d_in[11];
    float* out = (float*)d_out;

    const int BS = 4096, D = 2048, D3 = 6144;
    char* ws = (char*)d_ws;
    float*  qkv = (float*)ws;                     size_t off = (size_t)BS * D3 * 4;
    __bf16* qb  = (__bf16*)(ws + off);            off += (size_t)BS * D * 2;
    __bf16* kb  = (__bf16*)(ws + off);            off += (size_t)BS * D * 2;
    __bf16* vT  = (__bf16*)(ws + off);            off += (size_t)BS * D * 2;
    float*  ctx = (float*)(ws + off);

    dim3 g1(BS / GTM, D3 / GTN);
    gemm_nt_bf16x3<<<g1, 256, 0, stream>>>(x, Wqkv_w, Wqkv_b, qkv, BS, D3, D, 6.0f);

    ln_pack<<<BS, 256, 0, stream>>>(qkv, q_ln_g, q_ln_b, k_ln_g, k_ln_b, qb, kb, vT);

    dim3 ga(32, 32);   // (q tiles of 64, B*H)
    flash_attn<<<ga, 128, 0, stream>>>(qb, kb, vT, attn_bias, amask, qmask, ctx);

    dim3 g2(BS / GTM, D / GTN);
    gemm_nt_bf16x3<<<g2, 256, 0, stream>>>(ctx, out_w, out_b, out, BS, D, D, 0.0f);
}
